// FP8DynamicLinear_67774583931476
// MI455X (gfx1250) — 8-bit matrix operand order repaired and checked by reading, compiled, not run on hardware
//
#include <hip/hip_runtime.h>
#include <hip/hip_bf16.h>


typedef int   v16i __attribute__((ext_vector_type(16)));
typedef float v8f  __attribute__((ext_vector_type(8)));

#define FP8_MAX_F 448.0f

__device__ __forceinline__ float bf16_abs_f32(unsigned b) {
  return __uint_as_float((b & 0x7FFFu) << 16);
}
__device__ __forceinline__ float bf16_f32(unsigned b) {
  return __uint_as_float((b & 0xFFFFu) << 16);
}

__global__ void amax_x_kernel(const unsigned short* __restrict__ x, long long n,
                              unsigned* __restrict__ amax_bits) {
  const long long n4 = n >> 2;
  long long i = (long long)blockIdx.x * blockDim.x + threadIdx.x;
  const long long stride = (long long)gridDim.x * blockDim.x;
  const unsigned long long* p = (const unsigned long long*)x;
  float m = 0.0f;
  for (long long v = i; v < n4; v += stride) {
    unsigned long long u = p[v];
    m = fmaxf(m, bf16_abs_f32((unsigned)(u & 0xFFFFull)));
    m = fmaxf(m, bf16_abs_f32((unsigned)((u >> 16) & 0xFFFFull)));
    m = fmaxf(m, bf16_abs_f32((unsigned)((u >> 32) & 0xFFFFull)));
    m = fmaxf(m, bf16_abs_f32((unsigned)((u >> 48) & 0xFFFFull)));
  }
  if (i == 0)
    for (long long t = n4 << 2; t < n; ++t) m = fmaxf(m, bf16_abs_f32(x[t]));
#pragma unroll
  for (int off = 16; off; off >>= 1) m = fmaxf(m, __shfl_xor(m, off, 32));
  if ((threadIdx.x & 31) == 0) atomicMax(amax_bits, __float_as_uint(m));
}

__global__ void amax_w_kernel(const float* __restrict__ w, long long n,
                              unsigned* __restrict__ amax_bits) {
  const long long n4 = n >> 2;
  long long i = (long long)blockIdx.x * blockDim.x + threadIdx.x;
  const long long stride = (long long)gridDim.x * blockDim.x;
  const float4* p = (const float4*)w;
  float m = 0.0f;
  for (long long v = i; v < n4; v += stride) {
    float4 t = p[v];
    m = fmaxf(m, fmaxf(fmaxf(fabsf(t.x), fabsf(t.y)),
                       fmaxf(fabsf(t.z), fabsf(t.w))));
  }
  if (i == 0)
    for (long long t = n4 << 2; t < n; ++t) m = fmaxf(m, fabsf(w[t]));
#pragma unroll
  for (int off = 16; off; off >>= 1) m = fmaxf(m, __shfl_xor(m, off, 32));
  if ((threadIdx.x & 31) == 0) atomicMax(amax_bits, __float_as_uint(m));
}

__device__ __forceinline__ float clamp448(float f) {
  return fminf(fmaxf(f, -FP8_MAX_F), FP8_MAX_F);
}

__global__ void quant_x_kernel(const unsigned short* __restrict__ x,
                               unsigned* __restrict__ qx,
                               const unsigned* __restrict__ amax_bits,
                               long long n) {
  const float s = FP8_MAX_F / fmaxf(__uint_as_float(*amax_bits), 1e-12f);
  const long long n4 = n >> 2;
  long long i = (long long)blockIdx.x * blockDim.x + threadIdx.x;
  const long long stride = (long long)gridDim.x * blockDim.x;
  const unsigned long long* p = (const unsigned long long*)x;
  for (long long v = i; v < n4; v += stride) {
    unsigned long long u = p[v];
    float f0 = clamp448(bf16_f32((unsigned)(u        & 0xFFFFull)) * s);
    float f1 = clamp448(bf16_f32((unsigned)((u >> 16) & 0xFFFFull)) * s);
    float f2 = clamp448(bf16_f32((unsigned)((u >> 32) & 0xFFFFull)) * s);
    float f3 = clamp448(bf16_f32((unsigned)((u >> 48) & 0xFFFFull)) * s);
    int pk = 0;
    pk = __builtin_amdgcn_cvt_pk_fp8_f32(f0, f1, pk, false);
    pk = __builtin_amdgcn_cvt_pk_fp8_f32(f2, f3, pk, true);
    qx[v] = (unsigned)pk;
  }
  if (i == 0) {
    unsigned char* qb = (unsigned char*)qx;
    for (long long t = n4 << 2; t < n; ++t) {
      int pk = __builtin_amdgcn_cvt_pk_fp8_f32(clamp448(bf16_f32(x[t]) * s),
                                               0.0f, 0, false);
      qb[t] = (unsigned char)(pk & 0xFF);
    }
  }
}

__global__ void quant_w_kernel(const float* __restrict__ w,
                               unsigned* __restrict__ qw,
                               const unsigned* __restrict__ amax_bits,
                               long long n) {
  const float s = FP8_MAX_F / fmaxf(__uint_as_float(*amax_bits), 1e-12f);
  const long long n4 = n >> 2;
  long long i = (long long)blockIdx.x * blockDim.x + threadIdx.x;
  const long long stride = (long long)gridDim.x * blockDim.x;
  const float4* p = (const float4*)w;
  for (long long v = i; v < n4; v += stride) {
    float4 t = p[v];
    int pk = 0;
    pk = __builtin_amdgcn_cvt_pk_fp8_f32(clamp448(t.x * s), clamp448(t.y * s), pk, false);
    pk = __builtin_amdgcn_cvt_pk_fp8_f32(clamp448(t.z * s), clamp448(t.w * s), pk, true);
    qw[v] = (unsigned)pk;
  }
  if (i == 0) {
    unsigned char* qb = (unsigned char*)qw;
    for (long long t = n4 << 2; t < n; ++t) {
      int pk = __builtin_amdgcn_cvt_pk_fp8_f32(clamp448(w[t] * s), 0.0f, 0, false);
      qb[t] = (unsigned char)(pk & 0xFF);
    }
  }
}

__global__ __launch_bounds__(256)
void fp8_gemm_bias_kernel(const unsigned char* __restrict__ qx,
                          const unsigned char* __restrict__ qw,
                          const unsigned* __restrict__ amax_x_bits,
                          const unsigned* __restrict__ amax_w_bits,
                          const float* __restrict__ bias,
                          __hip_bfloat16* __restrict__ out,
                          int M, int N, int K) {
  const int lane  = threadIdx.x & 31;
  const int wave  = threadIdx.x >> 5;
  const int wm    = wave & 3;
  const int wn    = wave >> 2;
  const int lan16 = lane & 15;
  const int lhalf = lane >> 4;

  const int m0 = blockIdx.y * 128 + wm * 32;
  const int n0 = blockIdx.x * 128 + wn * 64;

  v8f acc[2][4];
#pragma unroll
  for (int a = 0; a < 2; ++a)
#pragma unroll
    for (int b = 0; b < 4; ++b)
      acc[a][b] = v8f{0.f, 0.f, 0.f, 0.f, 0.f, 0.f, 0.f, 0.f};

  const unsigned char* arow[2];
  arow[0] = qx + (size_t)(m0 + lan16) * K + 16 * lhalf;
  arow[1] = arow[0] + (size_t)16 * K;
  const unsigned char* brow[4];
  brow[0] = qw + (size_t)(n0 + lan16) * K + 16 * lhalf;
  brow[1] = brow[0] + (size_t)16 * K;
  brow[2] = brow[0] + (size_t)32 * K;
  brow[3] = brow[0] + (size_t)48 * K;

  for (int kb = 0; kb < K; kb += 128) {
    v16i A[2], B[4];
#pragma unroll
    for (int mt = 0; mt < 2; ++mt) {
      const unsigned char* ap = arow[mt] + kb;
#pragma unroll
      for (int half = 0; half < 2; ++half)
#pragma unroll
        for (int j = 0; j < 4; ++j) {
          int2 t = *(const int2*)(ap + 64 * half + 32 * (j >> 1) + 8 * (j & 1));
          A[mt][8 * half + 2 * j]     = t.x;
          A[mt][8 * half + 2 * j + 1] = t.y;
        }
    }
#pragma unroll
    for (int nt = 0; nt < 4; ++nt) {
      const unsigned char* bp = brow[nt] + kb;
#pragma unroll
      for (int g = 0; g < 4; ++g) {
        int4 t = *(const int4*)(bp + 32 * g);
        B[nt][4 * g + 0] = t.x;
        B[nt][4 * g + 1] = t.y;
        B[nt][4 * g + 2] = t.z;
        B[nt][4 * g + 3] = t.w;
      }
    }
#pragma unroll
    for (int mt = 0; mt < 2; ++mt)
#pragma unroll
      for (int nt = 0; nt < 4; ++nt)
        acc[mt][nt] = __builtin_amdgcn_wmma_f32_16x16x128_fp8_fp8(
            A[mt], B[nt], (short)0, acc[mt][nt], false, false);
  }

  const float sx = fmaxf(__uint_as_float(*amax_x_bits), 1e-12f) * (1.0f / FP8_MAX_F);
  const float sw = fmaxf(__uint_as_float(*amax_w_bits), 1e-12f) * (1.0f / FP8_MAX_F);
  const float s = sx * sw;

#pragma unroll
  for (int nt = 0; nt < 4; ++nt) {
    const int n = n0 + nt * 16 + lan16;
    const __hip_bfloat16 bb = __float2bfloat16(bias[n]);
#pragma unroll
    for (int mt = 0; mt < 2; ++mt) {
      const int mbase = m0 + mt * 16 + 8 * lhalf;
#pragma unroll
      for (int i = 0; i < 8; ++i) {
        __hip_bfloat16 r = __float2bfloat16(acc[mt][nt][i] * s);
        out[(size_t)(mbase + i) * N + n] = __hadd(r, bb);
      }
    }
  }
}

extern "C" void kernel_launch(void* const* d_in, const int* in_sizes, int n_in,
                              void* d_out, int out_size, void* d_ws, size_t ws_size,
                              hipStream_t stream) {
  const unsigned short* x  = (const unsigned short*)d_in[0];
  const float* wgt         = (const float*)d_in[1];
  const float* bias        = (const float*)d_in[2];

  const long long xn = in_sizes[0];
  const long long wn = in_sizes[1];
  const int N = in_sizes[2];
  const int K = (int)(wn / N);
  const int M = (int)(xn / K);

  unsigned char* ws = (unsigned char*)d_ws;
  unsigned* amax_x = (unsigned*)(ws + 0);
  unsigned* amax_w = (unsigned*)(ws + 4);
  unsigned char* qx = ws + 256;
  unsigned char* qw = qx + (size_t)M * K;

  hipMemsetAsync(ws, 0, 256, stream);

  amax_x_kernel<<<1024, 256, 0, stream>>>(x, xn, amax_x);
  amax_w_kernel<<<1024, 256, 0, stream>>>(wgt, wn, amax_w);

  quant_x_kernel<<<4096, 256, 0, stream>>>(x, (unsigned*)qx, amax_x, xn);
  quant_w_kernel<<<4096, 256, 0, stream>>>(wgt, (unsigned*)qw, amax_w, wn);

  dim3 grid(N / 128, M / 128);
  fp8_gemm_bias_kernel<<<grid, 256, 0, stream>>>(
      qx, qw, amax_x, amax_w, bias, (__hip_bfloat16*)d_out, M, N, K);
}
